// AggregateUpdate_15307263443166
// MI455X (gfx1250) — matrix operand order repaired and checked by reading, not compiled, not run on hardware
//
#include <hip/hip_runtime.h>

// ---------------------------------------------------------------------------
// GNN aggregate+MLP for MI455X (gfx1250, wave32, WMMA).
//   Phase 1: scatter-add (fp32 atomics, L2-resident accumulator)
//   Phase 2: fused mean-divide + concat + [256x256 GEMM + ReLU] + [128x256 GEMM]
//            using v_wmma_f32_16x16x32_bf16 with LDS-staged A fragments and
//            pre-packed B fragments.
// ---------------------------------------------------------------------------

#define FEAT   128   // F_NODE == F_EDGE
#define KIN    256   // F_NODE + F_EDGE
#define HID    256
#define OUTF   128

typedef __attribute__((ext_vector_type(16))) __bf16        v16bf;
typedef __attribute__((ext_vector_type(8)))  float         v8f;
typedef __attribute__((ext_vector_type(4)))  unsigned int  v4u;

union BFrag { v16bf v; v4u u[2]; };

__device__ __forceinline__ unsigned short f2bf(float f) {
    unsigned int u = __float_as_uint(f);
    u += 0x7FFFu + ((u >> 16) & 1u);           // round-to-nearest-even
    return (unsigned short)(u >> 16);
}
__device__ __forceinline__ unsigned int pk2(float a, float b) {
    return (unsigned int)f2bf(a) | ((unsigned int)f2bf(b) << 16);
}

// ---------------------------------------------------------------------------
__global__ void zero_kernel(float* p, int n4) {
    int i = blockIdx.x * blockDim.x + threadIdx.x;
    if (i < n4) {
        float4 z; z.x = 0.f; z.y = 0.f; z.z = 0.f; z.w = 0.f;
        ((float4*)p)[i] = z;
    }
}

// Pack W1 [HID x KIN] and W2 [OUTF x HID] fp32 row-major into bf16
// B-fragment order: per 32x16 (KxN) tile, element (k,n):
//   lane = n + 16*((k>>3)&1), within-lane bf16 index = (k&7) + 8*(k>>4)
// Packed linear index = ((nt*8 + kt)*32 + lane)*16 + idx16.
__global__ void pack_weights(const float* __restrict__ W1,
                             const float* __restrict__ W2,
                             unsigned short* __restrict__ w1p,
                             unsigned short* __restrict__ w2p) {
    int t = blockIdx.x * blockDim.x + threadIdx.x;   // 0 .. 98303
    const float* W; unsigned short* dst; int p;
    if (t < 16 * 8 * 512) { W = W1; dst = w1p; p = t; }
    else                  { W = W2; dst = w2p; p = t - 16 * 8 * 512; }
    int tile = p >> 9;            // (nt*8 + kt)
    int lane = (p >> 4) & 31;
    int idx  = p & 15;
    int nt = tile >> 3, kt = tile & 7;
    int nl = lane & 15;
    int kl = (idx & 7) + 8 * (lane >> 4) + 16 * (idx >> 3);
    dst[p] = f2bf(W[(size_t)(nt * 16 + nl) * KIN + kt * 32 + kl]);
}

// One edge per 128 lanes; fp32 atomics into L2-resident accumulator.
__global__ void scatter_kernel(const float* __restrict__ ea,
                               const long long* __restrict__ ei,
                               float* __restrict__ summed,
                               float* __restrict__ counts,
                               int n_edges) {
    int e = blockIdx.x * 2 + (threadIdx.x >> 7);
    int t = threadIdx.x & 127;
    if (e >= n_edges) return;
    int c = (int)ei[(size_t)n_edges + e];          // edge_index[1][e]
    atomicAdd(&summed[(size_t)c * FEAT + t], ea[(size_t)e * FEAT + t]);
    if (t == 0) atomicAdd(&counts[c], 1.0f);
}

// ---------------------------------------------------------------------------
// Fused MLP: 64 nodes / block, 256 threads = 8 waves.
// A-fragment swizzle for 16-bit 16x32 tiles (per ISA 7.12.2):
//   element (m, k): lane = m + 16*((k>>3)&1), idx16 = (k&7) + 8*(k>>4)
// LDS tile base (bf16 units): ((mt*8 + kt)*32 + lane)*16
// ---------------------------------------------------------------------------
__global__ void __launch_bounds__(256)
mlp_kernel(const float* __restrict__ x,
           const float* __restrict__ summed,
           const float* __restrict__ counts,
           const unsigned short* __restrict__ w1p,
           const unsigned short* __restrict__ w2p,
           const float* __restrict__ b1,
           const float* __restrict__ b2,
           float* __restrict__ out,
           int n_nodes) {
    __shared__ __align__(16) unsigned short hA [64 * 256]; // concat input, swizzled
    __shared__ __align__(16) unsigned short h1A[64 * 256]; // hidden, swizzled

    const int tid = threadIdx.x;

    // ---- Stage h = [x | agg] as bf16 A-fragments (4 threads/row, 64 cols each)
    {
        int r  = tid >> 2;                 // 0..63
        int c0 = (tid & 3) * 64;           // 0,64 -> x ; 128,192 -> agg
        long long node = (long long)blockIdx.x * 64 + r;
        bool valid = node < n_nodes;
        float scale = 1.0f;
        const float4* src4 = nullptr;
        if (valid) {
            if (c0 < 128) {
                src4 = (const float4*)(x + (size_t)node * FEAT + c0);
            } else {
                src4 = (const float4*)(summed + (size_t)node * FEAT + (c0 - 128));
                scale = 1.0f / fmaxf(counts[node], 1.0f);
            }
        }
        int mt = r >> 4, m = r & 15;
#pragma unroll
        for (int g = 0; g < 8; ++g) {      // 8 groups of 8 consecutive cols
            int c = c0 + g * 8;
            float f[8];
            if (valid) {
                float4 a = src4[g * 2], b = src4[g * 2 + 1];
                f[0] = a.x * scale; f[1] = a.y * scale; f[2] = a.z * scale; f[3] = a.w * scale;
                f[4] = b.x * scale; f[5] = b.y * scale; f[6] = b.z * scale; f[7] = b.w * scale;
            } else {
#pragma unroll
                for (int j = 0; j < 8; ++j) f[j] = 0.0f;
            }
            int kt = c >> 5, k0 = c & 31;                  // k0 in {0,8,16,24}
            int lane = m + 16 * ((k0 >> 3) & 1);
            int idx0 = (k0 >> 4) * 8;
            v4u pk;
            pk.x = pk2(f[0], f[1]); pk.y = pk2(f[2], f[3]);
            pk.z = pk2(f[4], f[5]); pk.w = pk2(f[6], f[7]);
            *(v4u*)&hA[(((mt * 8 + kt) * 32 + lane) << 4) + idx0] = pk; // ds_store_b128
        }
    }
    __syncthreads();

    const int wave = tid >> 5;
    const int lane = tid & 31;
    const int mt   = wave & 3;      // m-subtile of the 64-row block
    const int nh   = wave >> 2;     // n-range half
    const int nloc = lane & 15;     // C-matrix: N = lane%16
    const int half = lane >> 4;     // C-matrix: M = vgpr + 8*half

    // ---- Layer 1: h[64x256] x W1^T -> relu -> h1A (bf16, swizzled)
#pragma unroll
    for (int nti = 0; nti < 8; ++nti) {
        int nt = nh * 8 + nti;
        v8f acc = {};
#pragma unroll
        for (int kt = 0; kt < 8; ++kt) {
            BFrag A, B;
            const v4u* pa = (const v4u*)&hA[((mt * 8 + kt) * 32 + lane) << 4];
            A.u[0] = pa[0]; A.u[1] = pa[1];
            const v4u* pb = (const v4u*)(w1p + (size_t)(((nt * 8 + kt) << 5) + lane) * 16);
            B.u[0] = pb[0]; B.u[1] = pb[1];
            acc = __builtin_amdgcn_wmma_f32_16x16x32_bf16(
                false, A.v, false, B.v, (short)0, acc, false, false);
        }
        float bias = b1[nt * 16 + nloc];
        int cg   = nt * 16 + nloc;          // global hidden column
        int kt2  = cg >> 5, k2 = cg & 31;
        int idx  = (k2 & 7) + 8 * (k2 >> 4);
        int lsel = 16 * ((k2 >> 3) & 1);
#pragma unroll
        for (int i = 0; i < 8; ++i) {
            int m = i + 8 * half;
            float v = fmaxf(acc[i] + bias, 0.0f);
            h1A[(((mt * 8 + kt2) * 32 + (m + lsel)) << 4) + idx] = f2bf(v);
        }
    }
    __syncthreads();

    // ---- Layer 2: h1[64x256] x W2^T -> out (fp32)
#pragma unroll
    for (int nti = 0; nti < 4; ++nti) {
        int nt = nh * 4 + nti;
        v8f acc = {};
#pragma unroll
        for (int kt = 0; kt < 8; ++kt) {
            BFrag A, B;
            const v4u* pa = (const v4u*)&h1A[((mt * 8 + kt) * 32 + lane) << 4];
            A.u[0] = pa[0]; A.u[1] = pa[1];
            const v4u* pb = (const v4u*)(w2p + (size_t)(((nt * 8 + kt) << 5) + lane) * 16);
            B.u[0] = pb[0]; B.u[1] = pb[1];
            acc = __builtin_amdgcn_wmma_f32_16x16x32_bf16(
                false, A.v, false, B.v, (short)0, acc, false, false);
        }
        float bias = b2[nt * 16 + nloc];
#pragma unroll
        for (int i = 0; i < 8; ++i) {
            int m = i + 8 * half;
            long long node = (long long)blockIdx.x * 64 + mt * 16 + m;
            if (node < n_nodes)
                out[(size_t)node * OUTF + nt * 16 + nloc] = acc[i] + bias;
        }
    }
}

// ---------------------------------------------------------------------------
extern "C" void kernel_launch(void* const* d_in, const int* in_sizes, int n_in,
                              void* d_out, int out_size, void* d_ws, size_t ws_size,
                              hipStream_t stream) {
    const float*     x  = (const float*)d_in[0];
    const long long* ei = (const long long*)d_in[1];   // int64 [2, E]
    const float*     ea = (const float*)d_in[2];
    const float*     W1 = (const float*)d_in[3];
    const float*     b1 = (const float*)d_in[4];
    const float*     W2 = (const float*)d_in[5];
    const float*     b2 = (const float*)d_in[6];
    float* out = (float*)d_out;

    const int n_nodes = in_sizes[0] / FEAT;
    const int n_edges = in_sizes[2] / FEAT;

    // Workspace layout (16B aligned):
    //   summed [n_nodes*FEAT f32] | counts [n_nodes f32] | w1p [64K bf16] | w2p [32K bf16]
    float* summed = (float*)d_ws;
    float* counts = summed + (size_t)n_nodes * FEAT;
    unsigned short* w1p = (unsigned short*)(counts + n_nodes);
    unsigned short* w2p = w1p + 16 * 8 * 512;

    // Zero accumulators every call (graph replays re-run the atomics).
    size_t nzero = (size_t)n_nodes * FEAT + (size_t)n_nodes;   // multiple of 4
    int n4 = (int)(nzero / 4);
    zero_kernel<<<(n4 + 255) / 256, 256, 0, stream>>>(summed, n4);

    // Repack weights to bf16 B-fragment layout (65536 + 32768 elements).
    pack_weights<<<(16 * 8 * 512 + 8 * 8 * 512) / 256, 256, 0, stream>>>(W1, W2, w1p, w2p);

    // Scatter-add edges (2 edges per 256-thread block).
    scatter_kernel<<<(n_edges + 1) / 2, 256, 0, stream>>>(ea, ei, summed, counts, n_edges);

    // Fused mean + concat + MLP.
    mlp_kernel<<<(n_nodes + 63) / 64, 256, 0, stream>>>(
        x, summed, counts, w1p, w2p, b1, b2, out, n_nodes);
}
